// AttentionCompiled_50173807952102
// MI455X (gfx1250) — matrix operand order repaired and checked by reading, not compiled, not run on hardware
//
#include <hip/hip_runtime.h>

// Problem constants (B=8, N=1024, C=768, H=12, D=64)
#define Hh   12
#define Dh   64
#define Bb   8
#define Nn   1024
#define Cc   768
#define QSCALE 0.125f      // 64^-0.5
#define LNEPS  1e-5f
#define NEG_BIG (-1.0e30f)

#if defined(__HIP_DEVICE_COMPILE__) && \
    __has_builtin(__builtin_amdgcn_global_load_async_to_lds_b128) && \
    __has_builtin(__builtin_amdgcn_s_wait_asynccnt)
#define HAVE_ASYNC_LDS 1
#else
#define HAVE_ASYNC_LDS 0
#endif

typedef __attribute__((ext_vector_type(16))) __bf16          v16bf;
typedef __attribute__((ext_vector_type(8)))  float           v8f;
typedef __attribute__((ext_vector_type(8)))  unsigned short  ushort8;
typedef __attribute__((ext_vector_type(16))) unsigned short  ushort16;

#if HAVE_ASYNC_LDS
typedef __attribute__((ext_vector_type(4))) int v4i;
typedef __attribute__((address_space(1))) v4i* gv4i_p;   // global v4i*
typedef __attribute__((address_space(3))) v4i* lv4i_p;   // LDS v4i*
#endif

static __device__ __forceinline__ unsigned short f2bf(float f) {
  unsigned int u = __float_as_uint(f);
  u += 0x7FFFu + ((u >> 16) & 1u);          // round to nearest even
  return (unsigned short)(u >> 16);
}
static __device__ __forceinline__ float bf2f(unsigned int bits16) {
  return __uint_as_float(bits16 << 16);
}
static __device__ __forceinline__ ushort8 ld8(const unsigned short* p) {
  return *(const ushort8*)p;
}
static __device__ __forceinline__ v16bf mkfrag(ushort8 lo, ushort8 hi) {
  ushort16 t = __builtin_shufflevector(lo, hi, 0,1,2,3,4,5,6,7,8,9,10,11,12,13,14,15);
  return __builtin_bit_cast(v16bf, t);
}
#define WMMA_BF16(a,b,c) \
  __builtin_amdgcn_wmma_f32_16x16x32_bf16(false,(a),false,(b),(short)0,(c),false,false)

// A fragment: 16x32 bf16, A row-major [*, lda]. lane holds row (lane&15);
// halves h -> k = (h&7) + 8*(lane>>4) + 16*(h>>3)
static __device__ __forceinline__ v16bf load_afrag(const unsigned short* A, int lda,
                                                   int row0, int k0, int lane) {
  const unsigned short* p = A + (long)(row0 + (lane & 15)) * lda + k0 + 8 * (lane >> 4);
  return mkfrag(ld8(p), ld8(p + 16));
}

// 32x64 per-wave output tile: 8 WMMAs per 12 x b128 loads per K-step.
// Base pointers are hoisted so all steady-state addresses are base+imm offsets.
static __device__ __forceinline__ void wave_gemm_32x64(
    const unsigned short* A, int lda, int row0,
    const unsigned short* W, int ldb, int col0,
    int K, int lane, v8f acc[2][4]) {
  const unsigned short* pa0 = A + (long)(row0 + (lane & 15)) * lda + 8 * (lane >> 4);
  const unsigned short* pa1 = pa0 + (long)16 * lda;
  const unsigned short* pb  = W + (long)(col0 + (lane & 15)) * ldb + 8 * (lane >> 4);
  #pragma unroll 4
  for (int k0 = 0; k0 < K; k0 += 32) {
    v16bf a0 = mkfrag(ld8(pa0 + k0), ld8(pa0 + k0 + 16));
    v16bf a1 = mkfrag(ld8(pa1 + k0), ld8(pa1 + k0 + 16));
    #pragma unroll
    for (int j = 0; j < 4; ++j) {
      const unsigned short* q = pb + (long)j * 16 * ldb + k0;
      v16bf b = mkfrag(ld8(q), ld8(q + 16));
      acc[0][j] = WMMA_BF16(a0, b, acc[0][j]);
      acc[1][j] = WMMA_BF16(a1, b, acc[1][j]);
    }
  }
}

// ---------------- fp32 -> bf16 conversion ----------------
__global__ __launch_bounds__(256) void cvt_bf16_kernel(const float* __restrict__ in,
                                                       unsigned short* __restrict__ out,
                                                       int n) {
  int i = blockIdx.x * 256 + threadIdx.x;
  if (i < n) out[i] = f2bf(in[i]);
}

// ---------------- QKV GEMM: [8192,768] @ [2304,768]^T, scatter to q/k/v [B,H,N,D] bf16
// block tile 64x128 (4 waves, 2x2), wave tile 32x64 (exactly one head wide)
__global__ __launch_bounds__(128) void qkv_gemm_kernel(
    const unsigned short* __restrict__ xb,
    const unsigned short* __restrict__ wb,
    unsigned short* __restrict__ qbuf,
    unsigned short* __restrict__ kbuf,
    unsigned short* __restrict__ vbuf) {
  int tid = threadIdx.x, lane = tid & 31, w = tid >> 5;
  int row0 = blockIdx.x * 64 + (w >> 1) * 32;
  int col0 = blockIdx.y * 128 + (w & 1) * 64;
  v8f acc[2][4] = {};
  wave_gemm_32x64(xb, Cc, row0, wb, Cc, col0, Cc, lane, acc);

  int which = col0 / Cc;                 // 0:q 1:k 2:v (constant per wave)
  int h     = (col0 % Cc) >> 6;          // wave tile == one head
  unsigned short* dst = (which == 0) ? qbuf : ((which == 1) ? kbuf : vbuf);
  #pragma unroll
  for (int i = 0; i < 2; ++i)
    #pragma unroll
    for (int jt = 0; jt < 4; ++jt)
      #pragma unroll
      for (int r = 0; r < 8; ++r) {
        int m  = row0 + i * 16 + r + 8 * (lane >> 4);
        int b  = m >> 10, ns = m & (Nn - 1);
        dst[(((long)(b * Hh + h) * Nn) + ns) * Dh + jt * 16 + (lane & 15)] = f2bf(acc[i][jt][r]);
      }
}

// ---------------- per-head LayerNorm over D=64 (wave per row), q gets *QSCALE
__global__ __launch_bounds__(256) void ln_kernel(unsigned short* __restrict__ qbuf,
                                                 unsigned short* __restrict__ kbuf) {
  int tid = threadIdx.x, lane = tid & 31, w = tid >> 5;
  long row = (long)blockIdx.x * 8 + w;
  unsigned short* buf = (blockIdx.y == 0) ? qbuf : kbuf;
  float sc = (blockIdx.y == 0) ? QSCALE : 1.0f;
  unsigned int* p = (unsigned int*)(buf + row * Dh) + lane;
  unsigned int u = *p;
  float x0 = bf2f(u & 0xffffu), x1 = bf2f(u >> 16);
  float s = x0 + x1;
  #pragma unroll
  for (int m = 1; m < 32; m <<= 1) s += __shfl_xor(s, m, 32);
  float mean = s * (1.0f / 64.0f);
  float d0 = x0 - mean, d1 = x1 - mean;
  float ss = d0 * d0 + d1 * d1;
  #pragma unroll
  for (int m = 1; m < 32; m <<= 1) ss += __shfl_xor(ss, m, 32);
  float rstd = rsqrtf(ss * (1.0f / 64.0f) + LNEPS) * sc;
  *p = (unsigned int)f2bf(d0 * rstd) | ((unsigned int)f2bf(d1 * rstd) << 16);
}

// ---------------- flash attention: 4 waves x 16 query rows, 32-key blocks
__global__ __launch_bounds__(128) void flash_kernel(
    const unsigned short* __restrict__ qbuf,
    const unsigned short* __restrict__ kbuf,
    const unsigned short* __restrict__ vbuf,
    unsigned short* __restrict__ aob) {
  __shared__ __align__(16) unsigned short KT[32 * 64];     // K block, row-major (async-staged)
  __shared__ __align__(16) unsigned short VT[64 * 48];     // V block transposed: [d][kk], stride 48
  __shared__ __align__(16) unsigned short Pb[4][16 * 32];  // per-wave P scratch, stride 32

  int tid = threadIdx.x, lane = tid & 31, w = tid >> 5;
  int bh = blockIdx.x;
  int q0 = blockIdx.y * 64 + w * 16;
  const unsigned short* qh = qbuf + (long)bh * Nn * Dh;
  const unsigned short* kh = kbuf + (long)bh * Nn * Dh;
  const unsigned short* vh = vbuf + (long)bh * Nn * Dh;

  v16bf aq0 = load_afrag(qh, Dh, q0, 0, lane);
  v16bf aq1 = load_afrag(qh, Dh, q0, 32, lane);

  float mrow[8], lrow[8];
  v8f o[4] = {};
  #pragma unroll
  for (int r = 0; r < 8; ++r) { mrow[r] = NEG_BIG; lrow[r] = 0.0f; }

  int vr = tid >> 2;            // 0..31: row within the 32-key block
  int vc = (tid & 3) * 16;      // 16-half column group
  unsigned short* pp = Pb[w];

  for (int j = 0; j < Nn / 32; ++j) {
    __syncthreads();
    { // stage K block (32x64) into LDS, async when available
      const unsigned short* ksrc = kh + (long)(j * 32 + vr) * Dh + vc;
      unsigned short* kdst = &KT[vr * Dh + vc];
#if HAVE_ASYNC_LDS
      __builtin_amdgcn_global_load_async_to_lds_b128((gv4i_p)ksrc, (lv4i_p)kdst, 0, 0);
      __builtin_amdgcn_global_load_async_to_lds_b128((gv4i_p)ksrc, (lv4i_p)kdst, 16, 0);
#else
      *(ushort8*)kdst = ld8(ksrc);
      *(ushort8*)(kdst + 8) = ld8(ksrc + 8);
#endif
    }
    { // stage V block (32x64) transposed into LDS
      const unsigned short* src = vh + (long)(j * 32 + vr) * Dh + vc;
      ushort8 lo = ld8(src), hi = ld8(src + 8);
      #pragma unroll
      for (int e = 0; e < 8; ++e) {
        VT[(vc + e) * 48 + vr]     = lo[e];
        VT[(vc + 8 + e) * 48 + vr] = hi[e];
      }
    }
    if (j + 1 < Nn / 32) {  // prefetch next K/V blocks (global_prefetch_b8)
      __builtin_prefetch(kh + (long)(j + 1) * 32 * Dh + tid * 16, 0, 1);
      __builtin_prefetch(vh + (long)(j + 1) * 32 * Dh + tid * 16, 0, 1);
    }
#if HAVE_ASYNC_LDS
    __builtin_amdgcn_s_wait_asynccnt(0);
#endif
    __syncthreads();

    // S = Q K^T for this block's 32 keys (two 16x16 D-tiles), K frags from LDS
    v8f s0 = {}, s1 = {};
    #pragma unroll
    for (int t = 0; t < 2; ++t) {
      v8f* st = t ? &s1 : &s0;
      const unsigned short* kp0 = KT + (t * 16 + (lane & 15)) * Dh + 8 * (lane >> 4);
      v16bf bk0 = mkfrag(ld8(kp0), ld8(kp0 + 16));
      *st = WMMA_BF16(aq0, bk0, *st);
      v16bf bk1 = mkfrag(ld8(kp0 + 32), ld8(kp0 + 48));
      *st = WMMA_BF16(aq1, bk1, *st);
    }

    // online softmax; D-layout: row = r + 8*(lane>>4), col = lane&15
    #pragma unroll
    for (int r = 0; r < 8; ++r) {
      float mx = fmaxf(s0[r], s1[r]);
      #pragma unroll
      for (int msk = 1; msk < 16; msk <<= 1) mx = fmaxf(mx, __shfl_xor(mx, msk, 32));
      float mnew = fmaxf(mrow[r], mx);
      float corr = __expf(mrow[r] - mnew);
      float p0 = __expf(s0[r] - mnew);
      float p1 = __expf(s1[r] - mnew);
      float rs = p0 + p1;
      #pragma unroll
      for (int msk = 1; msk < 16; msk <<= 1) rs += __shfl_xor(rs, msk, 32);
      lrow[r] = lrow[r] * corr + rs;
      mrow[r] = mnew;
      #pragma unroll
      for (int c = 0; c < 4; ++c) o[c][r] *= corr;
      int m = r + 8 * (lane >> 4);
      pp[m * 32 + (lane & 15)]      = f2bf(p0);
      pp[m * 32 + 16 + (lane & 15)] = f2bf(p1);
    }

    // O += P V : A frag (16x32) from P scratch, B frags from transposed V tile
    {
      const unsigned short* ap = pp + (lane & 15) * 32 + 8 * (lane >> 4);
      v16bf aP = mkfrag(ld8(ap), ld8(ap + 16));
      #pragma unroll
      for (int c = 0; c < 4; ++c) {
        const unsigned short* bp = VT + (c * 16 + (lane & 15)) * 48 + 8 * (lane >> 4);
        v16bf bV = mkfrag(ld8(bp), ld8(bp + 16));
        o[c] = WMMA_BF16(aP, bV, o[c]);
      }
    }
  }

  // epilogue: normalize, write bf16 directly in [B*N, C] projection layout
  int b = bh / Hh, h = bh % Hh;
  #pragma unroll
  for (int r = 0; r < 8; ++r) {
    float inv = 1.0f / lrow[r];
    int ns = q0 + r + 8 * (lane >> 4);
    long row = (long)b * Nn + ns;
    #pragma unroll
    for (int c = 0; c < 4; ++c)
      aob[row * Cc + h * Dh + c * 16 + (lane & 15)] = f2bf(o[c][r] * inv);
  }
}

// ---------------- output projection: [8192,768] @ [768,768]^T -> fp32 out
__global__ __launch_bounds__(128) void proj_gemm_kernel(
    const unsigned short* __restrict__ aob,
    const unsigned short* __restrict__ wpb,
    float* __restrict__ out) {
  int tid = threadIdx.x, lane = tid & 31, w = tid >> 5;
  int row0 = blockIdx.x * 64 + (w >> 1) * 32;
  int col0 = blockIdx.y * 128 + (w & 1) * 64;
  v8f acc[2][4] = {};
  wave_gemm_32x64(aob, Cc, row0, wpb, Cc, col0, Cc, lane, acc);
  #pragma unroll
  for (int i = 0; i < 2; ++i)
    #pragma unroll
    for (int jt = 0; jt < 4; ++jt)
      #pragma unroll
      for (int r = 0; r < 8; ++r)
        out[(long)(row0 + i * 16 + r + 8 * (lane >> 4)) * Cc + col0 + jt * 16 + (lane & 15)]
            = acc[i][jt][r];
}

extern "C" void kernel_launch(void* const* d_in, const int* in_sizes, int n_in,
                              void* d_out, int out_size, void* d_ws, size_t ws_size,
                              hipStream_t stream) {
  (void)in_sizes; (void)n_in; (void)out_size; (void)ws_size;
  const float* x     = (const float*)d_in[0];
  const float* wqkv  = (const float*)d_in[1];
  const float* wproj = (const float*)d_in[2];
  float* out = (float*)d_out;

  const int M = Bb * Nn;            // 8192
  const int nX  = M * Cc;           // 6291456
  const int nWq = 3 * Cc * Cc;      // 1769472
  const int nWp = Cc * Cc;          // 589824
  const int nHd = Bb * Hh * Nn * Dh;// 6291456

  char* ws = (char*)d_ws;
  size_t off = 0;
  auto carve = [&](size_t nelem) -> unsigned short* {
    off = (off + 255) & ~(size_t)255;
    unsigned short* p = (unsigned short*)(ws + off);
    off += nelem * sizeof(unsigned short);
    return p;
  };
  unsigned short* xb  = carve(nX);
  unsigned short* wqb = carve(nWq);
  unsigned short* wpb = carve(nWp);
  unsigned short* qb  = carve(nHd);
  unsigned short* kb  = carve(nHd);
  unsigned short* vb  = carve(nHd);
  unsigned short* aob = carve(nX);

  cvt_bf16_kernel<<<(nX  + 255) / 256, 256, 0, stream>>>(x,     xb,  nX);
  cvt_bf16_kernel<<<(nWq + 255) / 256, 256, 0, stream>>>(wqkv,  wqb, nWq);
  cvt_bf16_kernel<<<(nWp + 255) / 256, 256, 0, stream>>>(wproj, wpb, nWp);

  qkv_gemm_kernel<<<dim3(M / 64, (3 * Cc) / 128), 128, 0, stream>>>(xb, wqb, qb, kb, vb);
  ln_kernel<<<dim3((Bb * Hh * Nn) / 8, 2), 256, 0, stream>>>(qb, kb);
  flash_kernel<<<dim3(Bb * Hh, Nn / 64), 128, 0, stream>>>(qb, kb, vb, aob);
  proj_gemm_kernel<<<dim3(M / 64, Cc / 128), 128, 0, stream>>>(aob, wpb, out);
}
